// CorrCosine_31293131718889
// MI455X (gfx1250) — matrix operand order repaired and checked by reading, compiled, not run on hardware
//
#include <hip/hip_runtime.h>
#include <hip/hip_bf16.h>

// ---------------------------------------------------------------------------
// CorrCosine on MI455X (gfx1250):
//   sim[b, pc, pr] = sum_c curN[b,c,pc] * refN[b,c,pr]   (L2-normalized on C)
// Batched GEMM, M=N=4096, K=256, B=8.  Output 512MB f32 -> HBM-store bound
// (~22us @ 23.3TB/s). Math: v_wmma_f32_16x16x32_f16 on normalized (|x|<=1)
// f16 data, f32 accumulation. Tile staging via Tensor Data Mover (TDM)
// tensor_load_to_lds with hardware row padding (TENSORcnt-tracked DMA).
// ---------------------------------------------------------------------------

typedef __attribute__((ext_vector_type(16))) _Float16 v16h;
typedef __attribute__((ext_vector_type(8)))  _Float16 v8h;
typedef __attribute__((ext_vector_type(8)))  float    v8f;
typedef __attribute__((ext_vector_type(4)))  unsigned int v4u;
typedef __attribute__((ext_vector_type(8)))  int      v8i;
typedef __attribute__((ext_vector_type(4)))  int      v4i;

#define B_   8
#define C_   256
#define P_   4096            // 64*64 pixels
#define TILE 128             // block output tile: TILE x TILE
#define LDS_STRIDE 264       // 256 + 8 halves pad (528B row -> 4-bank rotate)
#define SMEM_HALVES (2 * TILE * LDS_STRIDE)
#define SB_LDS_OFF  (TILE * LDS_STRIDE * 2)   // byte offset of B tile in LDS

// ---------------------------------------------------------------------------
// Kernel 1: L2-normalize along C, cast to f16, write transposed [B, P, C]
// so GEMM fragment loads are K-contiguous. One thread per pixel.
// ---------------------------------------------------------------------------
__global__ void corrcos_norm_t_kernel(const float* __restrict__ in,
                                      _Float16* __restrict__ out) {
    const int p = blockIdx.x * blockDim.x + threadIdx.x;   // pixel in [0,P)
    const int b = blockIdx.y;
    const float* src = in + (size_t)b * C_ * P_ + p;

    float ss = 0.f;
#pragma unroll 8
    for (int c = 0; c < C_; ++c) {
        float v = src[(size_t)c * P_];                     // coalesced per lane
        ss += v * v;
    }
    const float scale = 1.0f / fmaxf(sqrtf(ss), 1e-12f);

    _Float16* dst = out + ((size_t)b * P_ + p) * C_;       // 512B per thread
#pragma unroll
    for (int g = 0; g < C_; g += 8) {                      // 16B packed stores
        v8h pk;
#pragma unroll
        for (int j = 0; j < 8; ++j)
            pk[j] = (_Float16)(src[(size_t)(g + j) * P_] * scale);
        *(v8h*)(dst + g) = pk;                             // re-reads hit L2
    }
}

// ---------------------------------------------------------------------------
// TDM descriptor builder: DMA a 128-row x 512B tile (row stride 512B in
// global) into LDS at ldsOff, inserting 16B pad after every 512B row
// (pad_interval code 6 -> 128 DWORDs; pad_amount code 3 -> 4 DWORDs),
// matching LDS_STRIDE = 264 halves. data_size = 8B units.
// ---------------------------------------------------------------------------
__device__ __forceinline__ void tdm_load_tile(const _Float16* gsrc,
                                              unsigned ldsOff) {
    unsigned long long ga = (unsigned long long)(uintptr_t)gsrc;

    v4u g0;
    g0.x = 1u;                                        // count=1, user mode
    g0.y = ldsOff;                                    // lds_addr (bytes)
    g0.z = (unsigned)ga;                              // global_addr[31:0]
    g0.w = (unsigned)((ga >> 32) & 0x1FFFFFFull)      // global_addr[56:32]
           | (2u << 30);                              // type = 2 ("image")

    v8i g1;
    g1[0] = (3 << 16)        // data_size = 3 -> 8-byte units
          | (1 << 20)        // pad_enable
          | (6 << 22)        // pad_interval: 128 DWORDs (512B) per row
          | (3 << 25);       // pad_amount: 4 DWORDs (16B) pad
    g1[1] = (64 & 0xFFFF) << 16;        // [63:48] tensor_dim0 lo16 = 64 units
    g1[2] = (128 & 0xFFFF) << 16;       // [95:80+] tensor_dim1 lo16 = 128 rows
    g1[3] = (64 << 16);                 // [127:112] tile_dim0 = 64 units
    g1[4] = 128;                        // [143:128] tile_dim1 = 128 rows
    g1[5] = 64;                         // tensor_dim0_stride = 64 units (512B)
    g1[6] = 0;
    g1[7] = 0;

    v4i gz4 = {0, 0, 0, 0};             // 2-D tensor: groups 2/3 unused
    v8i gz8 = {0, 0, 0, 0, 0, 0, 0, 0};

    // 6-arg toolchain form: (g0, g1, g2, g3, extra, cpol)
    __builtin_amdgcn_tensor_load_to_lds(g0, g1, gz4, gz4, gz8, 0);
    __builtin_amdgcn_s_wait_tensorcnt(0);   // issuing wave waits for DMA done
}

// ---------------------------------------------------------------------------
// Kernel 2: batched GEMM via WMMA. Block = 256 threads (8 waves, wave32).
// Block tile 128x128; wave tile 32x64 (2 M-frags x 4 N-frags = 8 v8f accs).
// Full K=256 staged in LDS by TDM (2 x 128 x 264 halves = 132KB of 320KB).
// ---------------------------------------------------------------------------
__global__ void corrcos_wmma_kernel(const _Float16* __restrict__ Acur,  // [B,P,C]
                                    const _Float16* __restrict__ Bref,  // [B,P,C]
                                    float* __restrict__ out) {          // [B,P,P]
    extern __shared__ __align__(16) _Float16 smem[];
    _Float16* sA = smem;                     // 128 rows (p_cur) x 256 K, padded
    _Float16* sB = smem + TILE * LDS_STRIDE; // 128 rows (p_ref) x 256 K, padded

    const int tid = threadIdx.x;
    const int b  = blockIdx.z;
    const int m0 = blockIdx.y * TILE;        // cur-pixel tile base (rows)
    const int n0 = blockIdx.x * TILE;        // ref-pixel tile base (cols)
    const int w  = tid >> 5;                 // wave id 0..7

    // ---- stage global -> LDS with Tensor Data Mover (one DMA per tile).
    if (w == 0) {
        tdm_load_tile(Acur + ((size_t)b * P_ + m0) * C_, 0u);
    } else if (w == 1) {
        tdm_load_tile(Bref + ((size_t)b * P_ + n0) * C_, (unsigned)SB_LDS_OFF);
    }
    __syncthreads();                         // EXEC reconverges to all-1s

    const int lane = tid & 31;
    const int half = lane >> 4;       // half-wave select
    const int l16  = lane & 15;
    const int wm   = w >> 1;          // 0..3 -> 32-row stripe
    const int wn   = w & 1;           // 0..1 -> 64-col stripe
    const int mBase = wm * 32;
    const int nBase = wn * 64;

    v8f acc[2][4] = {};

    for (int k0 = 0; k0 < C_; k0 += 32) {
        v16h af[2], bf[4];
        // A frag 16x32 f16 (ISA 7.12.2): lane<16 K-blocks {0..7,16..23},
        // lane>=16 K-blocks {8..15,24..31}; M = lane%16.
#pragma unroll
        for (int mi = 0; mi < 2; ++mi) {
            const _Float16* pA =
                sA + (mBase + mi * 16 + l16) * LDS_STRIDE + k0 + half * 8;
            v8h lo = *(const v8h*)(pA);
            v8h hi = *(const v8h*)(pA + 16);
            af[mi] = __builtin_shufflevector(lo, hi, 0, 1, 2, 3, 4, 5, 6, 7,
                                             8, 9, 10, 11, 12, 13, 14, 15);
        }
        // B frag 32x16 f16: N = lane%16; lane<16 holds K {0..7,16..23},
        // lane>=16 holds K {8..15,24..31} (same K order as A).
#pragma unroll
        for (int ni = 0; ni < 4; ++ni) {
            const _Float16* pB =
                sB + (nBase + ni * 16 + l16) * LDS_STRIDE + k0 + half * 8;
            v8h lo = *(const v8h*)(pB);
            v8h hi = *(const v8h*)(pB + 16);
            bf[ni] = __builtin_shufflevector(lo, hi, 0, 1, 2, 3, 4, 5, 6, 7,
                                             8, 9, 10, 11, 12, 13, 14, 15);
        }
#pragma unroll
        for (int mi = 0; mi < 2; ++mi)
#pragma unroll
            for (int ni = 0; ni < 4; ++ni)
                acc[mi][ni] = __builtin_amdgcn_wmma_f32_16x16x32_f16(
                    /*neg_a=*/false, af[mi], /*neg_b=*/false, bf[ni],
                    /*c_mod=*/(short)0, acc[mi][ni],
                    /*reuse_a=*/false, /*reuse_b=*/false);
    }

    // ---- epilogue: C/D layout = VGPR v -> M = v + 8*half, N = lane%16.
    // Output is write-once 512MB stream -> non-temporal stores.
    const size_t outBase = (size_t)b * P_ * P_;
#pragma unroll
    for (int mi = 0; mi < 2; ++mi) {
        const int row0 = m0 + mBase + mi * 16 + half * 8;
#pragma unroll
        for (int ni = 0; ni < 4; ++ni) {
            const int col = n0 + nBase + ni * 16 + l16;
#pragma unroll
            for (int v = 0; v < 8; ++v) {
                __builtin_nontemporal_store(
                    acc[mi][ni][v],
                    out + outBase + (size_t)(row0 + v) * P_ + col);
            }
        }
    }
}

// ---------------------------------------------------------------------------
extern "C" void kernel_launch(void* const* d_in, const int* in_sizes, int n_in,
                              void* d_out, int out_size, void* d_ws, size_t ws_size,
                              hipStream_t stream) {
    // setup_inputs order: ref_features, cur_features (both f32 [8,256,64,64])
    const float* ref = (const float*)d_in[0];
    const float* cur = (const float*)d_in[1];
    float* out = (float*)d_out;

    // Workspace: two normalized-transposed f16 tensors [B, P, C] (16MB each).
    _Float16* curT = (_Float16*)d_ws;
    _Float16* refT = curT + (size_t)B_ * P_ * C_;

    dim3 nGrid(P_ / 256, B_);
    corrcos_norm_t_kernel<<<nGrid, 256, 0, stream>>>(cur, curT);
    corrcos_norm_t_kernel<<<nGrid, 256, 0, stream>>>(ref, refT);

    dim3 gGrid(P_ / TILE, P_ / TILE, B_);   // 32 x 32 x 8 blocks
    size_t smemBytes = (size_t)SMEM_HALVES * sizeof(_Float16);  // 132KB
    corrcos_wmma_kernel<<<gGrid, 256, smemBytes, stream>>>(curT, refT, out);
}
